// VQEMA_26096221290584
// MI455X (gfx1250) — matrix operand order repaired and checked by reading, not compiled, not run on hardware
//
#include <hip/hip_runtime.h>
#include <hip/hip_bf16.h>

// ---------------- problem constants ----------------
#define EMB_K   512
#define EMB_D   256
#define BATCH   16
#define TLEN    4096
#define NVEC    (BATCH * TLEN)          // 65536 vectors
#define GAMMA_  0.99f
#define BETA_   0.25f
#define EPS_    1e-5f

// ---------------- vector types (POD, union-safe) ----------------
typedef __attribute__((ext_vector_type(16))) _Float16 v16h;
typedef __attribute__((ext_vector_type(8)))  float    v8f;
typedef __attribute__((ext_vector_type(4)))  float    f4;
typedef __attribute__((ext_vector_type(2)))  float    f2v;
typedef __attribute__((ext_vector_type(4)))  unsigned int u4;

union HFrag { v16h v; u4 u[2]; _Float16 h[16]; };

// ---------------- workspace layout (bytes) ----------------
// xT   : [NVEC, D] fp32 transposed inputs
// ebh  : [K, D] fp16 codebook
// cbuf : [K] fp32  0.5*||e||^2
// idx  : [NVEC] int argmin
// counts + dw + lossAcc : zeroed each call (contiguous)
// newW : [K, D] fp32 updated codebook
static const size_t XT_OFF     = 0;
static const size_t EBH_OFF    = (size_t)NVEC * EMB_D * 4;            // 67,108,864
static const size_t C_OFF      = EBH_OFF + (size_t)EMB_K * EMB_D * 2; // +262,144
static const size_t IDX_OFF    = C_OFF + EMB_K * 4;                   // +2048
static const size_t COUNTS_OFF = IDX_OFF + (size_t)NVEC * 4;          // +262,144
static const size_t DW_OFF     = COUNTS_OFF + EMB_K * 4;              // +2048
static const size_t LOSS_OFF   = DW_OFF + (size_t)EMB_K * EMB_D * 4;  // +524,288
static const size_t NEWW_OFF   = LOSS_OFF + 256;                      // pad
static const size_t ZERO_FLOATS = (EMB_K * 4 + (size_t)EMB_K * EMB_D * 4 + 256) / 4; // 131,648

// LDS layout for the argmin GEMM: padded f16 codebook rows (+8 halfs) + c[]
#define LDS_ROW_BYTES 528                       // 256*2 + 16 (bank-conflict pad)
#define LDS_C_OFF     (EMB_K * LDS_ROW_BYTES)   // 270,336
#define LDS_TOTAL     (LDS_C_OFF + EMB_K * 4)   // 272,384 B (<320 KB/WGP on CDNA5)

// =====================================================================
// K0: zero scratch (counts, dw, loss accumulator)
// =====================================================================
__global__ void k_zero(float* __restrict__ p, int n) {
    int i = blockIdx.x * blockDim.x + threadIdx.x;
    if (i < n) p[i] = 0.f;
}

// =====================================================================
// K1: transpose inputs [B, D, T] fp32 -> xT [B*T, D] fp32 (coalesced both ways)
// =====================================================================
__global__ void k_transpose(const float* __restrict__ in, float* __restrict__ xT) {
    __shared__ float tile[32][33];
    const int b  = blockIdx.z;
    const int t0 = blockIdx.x * 32;
    const int d0 = blockIdx.y * 32;
    const float* src = in + (size_t)b * EMB_D * TLEN;
    for (int j = threadIdx.y; j < 32; j += 8)
        tile[j][threadIdx.x] = src[(size_t)(d0 + j) * TLEN + t0 + threadIdx.x];
    __syncthreads();
    float* dst = xT + ((size_t)b * TLEN + t0) * EMB_D + d0;
    for (int j = threadIdx.y; j < 32; j += 8)
        dst[(size_t)j * EMB_D + threadIdx.x] = tile[threadIdx.x][j];
}

// =====================================================================
// K2: codebook -> fp16 + c[k] = 0.5*||e_k||^2
// =====================================================================
__global__ void k_prep_codes(const float* __restrict__ emb,
                             _Float16* __restrict__ ebh,
                             float* __restrict__ cbuf) {
    __shared__ float red[256];
    const int k = blockIdx.x, tid = threadIdx.x;
    float v = emb[(size_t)k * EMB_D + tid];
    ebh[(size_t)k * EMB_D + tid] = (_Float16)v;
    red[tid] = v * v;
    __syncthreads();
    for (int s = 128; s > 0; s >>= 1) { if (tid < s) red[tid] += red[tid + s]; __syncthreads(); }
    if (tid == 0) cbuf[k] = 0.5f * red[0];
}

// =====================================================================
// K3: WMMA distance GEMM + per-row argmin + counts
//   score(r,k) = x_r . e_k  via v_wmma_f32_16x16x32_f16
//   minimize   0.5*||e_k||^2 - score   (||x||^2 is row-constant)
// Full f16 codebook (512x264 halfs = 264KB) staged in dynamic LDS.
// =====================================================================
__global__ void __launch_bounds__(256)
k_argmin(const float* __restrict__ xT, const _Float16* __restrict__ ebh,
         const float* __restrict__ cbuf, int* __restrict__ idxb,
         float* __restrict__ counts) {
    extern __shared__ char smem[];
    float* csh = (float*)(smem + LDS_C_OFF);

    const int tid  = threadIdx.x;
    const int lane = tid & 31;
    const int wave = tid >> 5;
    const int l15  = lane & 15;
    const bool hi  = lane >= 16;

    // ---- stage f16 codebook into padded LDS rows (16B chunks, coalesced) ----
    {
        const u4* ebg = (const u4*)ebh;                // K*D/8 = 16384 chunks
        for (int ch = tid; ch < (EMB_K * EMB_D) / 8; ch += 256) {
            int row = ch >> 5;                         // 32 chunks per row
            int c8  = ch & 31;
            *(u4*)(smem + (size_t)row * LDS_ROW_BYTES + c8 * 16) = ebg[ch];
        }
        for (int k = tid; k < EMB_K; k += 256) csh[k] = cbuf[k];
    }
    __syncthreads();

    for (int slab = blockIdx.x * 128; slab < NVEC; slab += gridDim.x * 128) {
        const int row0 = slab + wave * 16;

        // prefetch next slab's A rows (CDNA5 global_prefetch_b8)
        if (slab + (int)gridDim.x * 128 < NVEC)
            __builtin_prefetch(xT + ((size_t)row0 + gridDim.x * 128 + l15) * EMB_D, 0, 0);

        // ---- load & convert A tiles: 16 rows x 256 K, fp32 -> f16 ----
        // A layout (16-bit A 16x32): lanes 0-15 row=l, K=[k0..k0+7],[k0+16..k0+23]
        //                            lanes 16-31 row=l-16, k0 += 8
        HFrag A[8];
        const float* xrow = xT + ((size_t)row0 + l15) * EMB_D;
        #pragma unroll
        for (int ks = 0; ks < 8; ++ks) {
            const int k0 = ks * 32 + (hi ? 8 : 0);
            f4 r0 = *(const f4*)(xrow + k0);
            f4 r1 = *(const f4*)(xrow + k0 + 4);
            f4 r2 = *(const f4*)(xrow + k0 + 16);
            f4 r3 = *(const f4*)(xrow + k0 + 20);
            A[ks].h[0]  = (_Float16)r0.x; A[ks].h[1]  = (_Float16)r0.y;
            A[ks].h[2]  = (_Float16)r0.z; A[ks].h[3]  = (_Float16)r0.w;
            A[ks].h[4]  = (_Float16)r1.x; A[ks].h[5]  = (_Float16)r1.y;
            A[ks].h[6]  = (_Float16)r1.z; A[ks].h[7]  = (_Float16)r1.w;
            A[ks].h[8]  = (_Float16)r2.x; A[ks].h[9]  = (_Float16)r2.y;
            A[ks].h[10] = (_Float16)r2.z; A[ks].h[11] = (_Float16)r2.w;
            A[ks].h[12] = (_Float16)r3.x; A[ks].h[13] = (_Float16)r3.y;
            A[ks].h[14] = (_Float16)r3.z; A[ks].h[15] = (_Float16)r3.w;
        }

        float mv[8]; int mi[8];
        #pragma unroll
        for (int m = 0; m < 8; ++m) { mv[m] = 3.4e38f; mi[m] = 0; }

        // ---- sweep 32 column tiles of 16 codes, running argmin ----
        for (int ct = 0; ct < 32; ++ct) {
            v8f acc = {};
            // B layout (16-bit B 32x16): lane col=l15; lanes 0-15 K=0..7,16..23,
            // lanes 16-31 K=8..15,24..31 (two 16B chunks per lane per K-step)
            const char* brow = smem + (size_t)(ct * 16 + l15) * LDS_ROW_BYTES + (hi ? 16 : 0);
            #pragma unroll
            for (int ks = 0; ks < 8; ++ks) {
                HFrag Bf;
                Bf.u[0] = *(const u4*)(brow + ks * 64);
                Bf.u[1] = *(const u4*)(brow + ks * 64 + 32);
                acc = __builtin_amdgcn_wmma_f32_16x16x32_f16(
                    false, A[ks].v, false, Bf.v, (short)0, acc, false, false);
            }
            const int   col = ct * 16 + l15;
            const float cc  = csh[col];
            #pragma unroll
            for (int m = 0; m < 8; ++m) {
                float dv = cc - acc[m];               // 0.5||e||^2 - x.e
                if (dv < mv[m]) { mv[m] = dv; mi[m] = col; }
            }
        }

        // ---- reduce argmin across the 16 lanes sharing each row ----
        #pragma unroll
        for (int m = 0; m < 8; ++m) {
            for (int off = 1; off < 16; off <<= 1) {
                float ov = __shfl_xor(mv[m], off, 16);
                int   oi = __shfl_xor(mi[m], off, 16);
                if (ov < mv[m] || (ov == mv[m] && oi < mi[m])) { mv[m] = ov; mi[m] = oi; }
            }
        }
        // lane 0 holds rows 0..7 (M=v), lane 16 holds rows 8..15 (M=8+v)
        if (l15 == 0) {
            const int rbase = row0 + (hi ? 8 : 0);
            #pragma unroll
            for (int m = 0; m < 8; ++m) {
                idxb[rbase + m] = mi[m];
                atomicAdd(counts + mi[m], 1.0f);
            }
        }
    }
}

// =====================================================================
// K4: dw[k] += x_n  (segment sum via fp32 atomics; coalesced x reads)
// =====================================================================
__global__ void k_dw(const float* __restrict__ xT, const int* __restrict__ idxb,
                     float* __restrict__ dw) {
    const size_t i = (size_t)blockIdx.x * blockDim.x + threadIdx.x; // N*D/4 threads
    const size_t e = i * 4;
    const int n = (int)(e >> 8);
    const int d = (int)(e & 255);
    const int k = idxb[n];
    f4 x = *(const f4*)(xT + e);
    float* p = dw + (size_t)k * EMB_D + d;
    atomicAdd(p + 0, x.x); atomicAdd(p + 1, x.y);
    atomicAdd(p + 2, x.z); atomicAdd(p + 3, x.w);
}

// =====================================================================
// K5: EMA update -> newW, plus perplexity PP (single block, 512 threads)
// =====================================================================
__global__ void k_update(const float* __restrict__ counts,
                         const float* __restrict__ ema_cs,
                         const float* __restrict__ ema_w,
                         const float* __restrict__ dw,
                         float* __restrict__ newW,
                         float* __restrict__ outPP) {
    __shared__ float ncs_s[EMB_K];
    __shared__ float red[EMB_K];
    const int tid = threadIdx.x;
    const float cnt = counts[tid];
    const float ncs = ema_cs[tid] * GAMMA_ + (1.f - GAMMA_) * cnt;

    red[tid] = ncs; __syncthreads();
    for (int s = 256; s > 0; s >>= 1) { if (tid < s) red[tid] += red[tid + s]; __syncthreads(); }
    const float n = red[0];
    __syncthreads();

    ncs_s[tid] = (ncs + EPS_) / (n + EMB_K * EPS_) * n;

    // perplexity: PP = exp(-sum p*log(p+1e-10)), p = counts/N
    const float p = cnt / (float)NVEC;
    red[tid] = p * logf(p + 1e-10f);
    __syncthreads();
    for (int s = 256; s > 0; s >>= 1) { if (tid < s) red[tid] += red[tid + s]; __syncthreads(); }
    if (tid == 0) outPP[0] = expf(-red[0]);

    for (int e = tid; e < EMB_K * EMB_D; e += 512) {
        int k = e >> 8;
        newW[e] = (ema_w[e] * GAMMA_ + (1.f - GAMMA_) * dw[e]) / ncs_s[k];
    }
}

// =====================================================================
// K6: Q = newW[idx[n]] ; write Q_st (== Q forward) transposed to [B,D,T];
//     accumulate sum of (Q-x)^2 for the loss
// =====================================================================
__global__ void k_quant(const float* __restrict__ xT, const int* __restrict__ idxb,
                        const float* __restrict__ newW, float* __restrict__ outQ,
                        float* __restrict__ lossAcc) {
    __shared__ float tile[32][33];
    __shared__ float red[256];
    const int b  = blockIdx.z;
    const int t0 = blockIdx.x * 32;
    const int d0 = blockIdx.y * 32;

    float acc = 0.f;
    for (int i = threadIdx.y; i < 32; i += 8) {
        const int n = b * TLEN + t0 + i;
        const int k = idxb[n];
        const int d = d0 + threadIdx.x;
        float q  = newW[(size_t)k * EMB_D + d];
        float x  = xT[(size_t)n * EMB_D + d];
        float df = q - x;
        acc += df * df;
        tile[i][threadIdx.x] = q;
    }
    __syncthreads();
    float* dst = outQ + (size_t)b * EMB_D * TLEN;
    for (int j = threadIdx.y; j < 32; j += 8)
        dst[(size_t)(d0 + j) * TLEN + t0 + threadIdx.x] = tile[threadIdx.x][j];

    const int tid = threadIdx.y * 32 + threadIdx.x;
    red[tid] = acc; __syncthreads();
    for (int s = 128; s > 0; s >>= 1) { if (tid < s) red[tid] += red[tid + s]; __syncthreads(); }
    if (tid == 0) atomicAdd(lossAcc, red[0]);
}

// =====================================================================
// K7: one-hot enc [N, K] (float2 stores: enc base is only 8B-aligned in d_out)
// =====================================================================
__global__ void k_enc(const int* __restrict__ idxb, float* __restrict__ enc) {
    const size_t i = (size_t)blockIdx.x * blockDim.x + threadIdx.x; // N*K/4 threads
    const size_t e = i * 4;
    const int n  = (int)(e >> 9);
    const int k  = (int)(e & 511);
    const int id = idxb[n];
    f2v a, b2;
    a.x  = (float)(k     == id); a.y  = (float)(k + 1 == id);
    b2.x = (float)(k + 2 == id); b2.y = (float)(k + 3 == id);
    *(f2v*)(enc + e)     = a;
    *(f2v*)(enc + e + 2) = b2;
}

// =====================================================================
// K8: finalize loss = BETA * sum / (N*D)
// =====================================================================
__global__ void k_finish(const float* __restrict__ lossAcc, float* __restrict__ out0) {
    out0[0] = BETA_ * lossAcc[0] / (float)((size_t)NVEC * EMB_D);
}

// =====================================================================
extern "C" void kernel_launch(void* const* d_in, const int* in_sizes, int n_in,
                              void* d_out, int out_size, void* d_ws, size_t ws_size,
                              hipStream_t stream) {
    const float* inputs = (const float*)d_in[0];   // [B, D, T]
    const float* emb    = (const float*)d_in[1];   // [K, D]
    const float* ema_w  = (const float*)d_in[2];   // [K, D]
    const float* ema_cs = (const float*)d_in[3];   // [K]

    float* out    = (float*)d_out;
    float* outQ   = out + 1;                                   // [B, D, T]
    float* outPP  = out + 1 + (size_t)BATCH * EMB_D * TLEN;    // scalar
    float* outEnc = outPP + 1;                                 // [N, K]

    char* ws = (char*)d_ws;
    float*    xT     = (float*)(ws + XT_OFF);
    _Float16* ebh    = (_Float16*)(ws + EBH_OFF);
    float*    cbuf   = (float*)(ws + C_OFF);
    int*      idxb   = (int*)(ws + IDX_OFF);
    float*    counts = (float*)(ws + COUNTS_OFF);
    float*    dw     = (float*)(ws + DW_OFF);
    float*    loss   = (float*)(ws + LOSS_OFF);
    float*    newW   = (float*)(ws + NEWW_OFF);

    // K0: zero counts+dw+lossAcc
    k_zero<<<(int)((ZERO_FLOATS + 255) / 256), 256, 0, stream>>>(counts, (int)ZERO_FLOATS);
    // K1: transpose inputs
    k_transpose<<<dim3(TLEN / 32, EMB_D / 32, BATCH), dim3(32, 8), 0, stream>>>(inputs, xT);
    // K2: f16 codebook + 0.5||e||^2
    k_prep_codes<<<EMB_K, 256, 0, stream>>>(emb, ebh, cbuf);
    // K3: WMMA GEMM + argmin + counts (272KB dynamic LDS)
    k_argmin<<<256, 256, LDS_TOTAL, stream>>>(xT, ebh, cbuf, idxb, counts);
    // K4: dw segment sum
    k_dw<<<(NVEC * EMB_D / 4) / 256, 256, 0, stream>>>(xT, idxb, dw);
    // K5: EMA update + PP
    k_update<<<1, EMB_K, 0, stream>>>(counts, ema_cs, ema_w, dw, newW, outPP);
    // K6: quantize + transpose out + loss partials
    k_quant<<<dim3(TLEN / 32, EMB_D / 32, BATCH), dim3(32, 8), 0, stream>>>(xT, idxb, newW, outQ, loss);
    // K7: one-hot encodings
    k_enc<<<(int)(((size_t)NVEC * EMB_K / 4) / 256), 256, 0, stream>>>(idxb, outEnc);
    // K8: loss scalar
    k_finish<<<1, 1, 0, stream>>>(loss, out);
}
